// FNO2d1_7559142441111
// MI455X (gfx1250) — compile-verified, matrix operand order checked by reading, not run on hardware
//
#include <hip/hip_runtime.h>
#include <math.h>

// ---------------------------------------------------------------------------
// FNO2d for MI455X (gfx1250): everything expressed as bf16-WMMA GEMMs.
// Truncated rfft2/irfft2 are small DFT matmuls (only 16x16 modes kept), so no
// FFT is needed. Activations/weights live in fp32 in HBM; tiles convert to
// bf16 in LDS and feed v_wmma_f32_16x16x32_bf16 with fp32 accumulation.
// All buffers < 2^31 elements -> 32-bit offset math throughout.
// Workspace requirement: ~164M floats (~630 MB) in d_ws.
// ---------------------------------------------------------------------------

typedef __attribute__((ext_vector_type(16))) __bf16 v16bf;
typedef __attribute__((ext_vector_type(8)))  __bf16 v8bf;
typedef __attribute__((ext_vector_type(4)))  __bf16 v4bf;
typedef __attribute__((ext_vector_type(8)))  float  v8f;

#define GEMM_BM 128
#define GEMM_BN 64
#define GEMM_BK 64   // two WMMA k-steps per LDS stage
#define APITCH  72   // pitch in bf16 elements; 72*2B=144B keeps 16B alignment

static __device__ __forceinline__ float gelu_f(float v) {
  return 0.5f * v * (1.0f + erff(v * 0.70710678118654752f));
}

static __device__ __forceinline__ v8f wmma_bf16(v16bf a, v16bf b, v8f c) {
  return __builtin_amdgcn_wmma_f32_16x16x32_bf16(false, a, false, b,
                                                 (short)0, c, false, false);
}

// bf16 WMMA NEG bits only negate C, so negate fragments via sign-bit XOR.
static __device__ __forceinline__ v16bf neg_frag(v16bf x) {
  union { v16bf v; unsigned int u[8]; } t;
  t.v = x;
#pragma unroll
  for (int i = 0; i < 8; ++i) t.u[i] ^= 0x80008000u;
  return t.v;
}

// A/B fragment per ISA 16-bit layout: lanes 0-15 take K {kb..kb+7,kb+16..kb+23}
// with kb=0, lanes 16-31 the same with kb=8. Two 16B LDS loads per fragment.
static __device__ __forceinline__ v16bf ld_frag(const __bf16* rowbase, int kb) {
  union { v16bf v; v8bf h[2]; } u;
  u.h[0] = *(const v8bf*)(rowbase + kb);
  u.h[1] = *(const v8bf*)(rowbase + kb + 16);
  return u.v;
}

static __device__ __forceinline__ v4bf cvt4(float4 v) {
  v4bf t;
  t[0] = (__bf16)v.x; t[1] = (__bf16)v.y; t[2] = (__bf16)v.z; t[3] = (__bf16)v.w;
  return t;
}

struct GemmParams {
  const float* A;  const float* A2;       // A2: optional second chain (same strides)
  int sAm, sAk;                           // element strides of A
  int bAdiv, bAhi, bAlo;                  // batch base = (z/div)*hi + (z%div)*lo
  const float* B;  const float* B2;       // B2 optional (same strides as B)
  int sBk, sBn;
  float* C;
  int sCm, sCn;
  int bCdiv, bChi, bClo;
  const float* bias;  const float* bias2; // indexed by n, optional
  int M, N, K;
  int flags;                              // 1 = GELU epilogue, 2 = negate A2 chain
};

// acc = A*B (+/- A2*B2); C = gelu?(acc + bias + bias2)
__global__ __launch_bounds__(256) void gemm_bf16_kernel(GemmParams p) {
  __shared__ __attribute__((aligned(16))) __bf16 sA [GEMM_BM * APITCH];
  __shared__ __attribute__((aligned(16))) __bf16 sA2[GEMM_BM * APITCH];
  __shared__ __attribute__((aligned(16))) __bf16 sB [GEMM_BN * APITCH];
  __shared__ __attribute__((aligned(16))) __bf16 sB2[GEMM_BN * APITCH];

  const int tid = threadIdx.x;
  const int z = blockIdx.z;
  const int abase = (z / p.bAdiv) * p.bAhi + (z % p.bAdiv) * p.bAlo;
  const int cbase = (z / p.bCdiv) * p.bChi + (z % p.bCdiv) * p.bClo;
  const int m0 = blockIdx.y * GEMM_BM;
  const int n0 = blockIdx.x * GEMM_BN;

  const bool hasA2 = (p.A2 != nullptr);
  const bool neg2  = (p.flags & 2) != 0;
  const bool kfA   = (p.sAk == 1);
  const bool kfB   = (p.sBk == 1);
  const bool fullM = (m0 + GEMM_BM <= p.M);
  const bool fullN = (n0 + GEMM_BN <= p.N);

  const int wave = tid >> 5;
  const int lane = tid & 31;
  const int half = lane >> 4;
  const int l15  = lane & 15;
  const int wm   = (wave & 3) * 32;   // 4 waves along M
  const int wn   = (wave >> 2) * 32;  // 2 waves along N
  const int kb   = half * 8;

  v8f acc[2][2];
#pragma unroll
  for (int a = 0; a < 2; ++a)
#pragma unroll
    for (int b = 0; b < 2; ++b)
#pragma unroll
      for (int r = 0; r < 8; ++r) acc[a][b][r] = 0.0f;

  for (int k0 = 0; k0 < p.K; k0 += GEMM_BK) {
    const bool fullK = (k0 + GEMM_BK <= p.K);
    __syncthreads();
    if (k0 + GEMM_BK < p.K)  // speculative prefetch of next A tile
      __builtin_prefetch((const void*)(p.A + abase + m0 * p.sAm +
                                       (k0 + GEMM_BK) * p.sAk), 0, 1);

    // ---------------- stage A (+A2): fp32 -> bf16 LDS ----------------
    if (fullM && fullK && kfA) {
      // k-contiguous: float4 loads, v4bf LDS stores
      const float* ga  = p.A + abase + m0 * p.sAm + k0;
      const float* ga2 = hasA2 ? (p.A2 + abase + m0 * p.sAm + k0) : nullptr;
#pragma unroll
      for (int f = tid; f < (GEMM_BM * GEMM_BK) / 4; f += 256) {
        const int i = f >> 4, j4 = (f & 15) << 2;     // 16 float4 per row
        const int go = i * p.sAm + j4;
        *(v4bf*)(sA + i * APITCH + j4) = cvt4(*(const float4*)(ga + go));
        if (hasA2) *(v4bf*)(sA2 + i * APITCH + j4) = cvt4(*(const float4*)(ga2 + go));
      }
    } else if (fullM && fullK && p.sAm == 1) {
      // m-contiguous: float4 along m, 4 scalar LDS stores
      const float* ga  = p.A + abase + m0 + k0 * p.sAk;
      const float* ga2 = hasA2 ? (p.A2 + abase + m0 + k0 * p.sAk) : nullptr;
#pragma unroll
      for (int f = tid; f < (GEMM_BM / 4) * GEMM_BK; f += 256) {
        const int i4 = (f & 31) << 2, j = f >> 5;     // 32 float4 per column
        const int go = i4 + j * p.sAk;
        const float4 v = *(const float4*)(ga + go);
        sA[(i4 + 0) * APITCH + j] = (__bf16)v.x;
        sA[(i4 + 1) * APITCH + j] = (__bf16)v.y;
        sA[(i4 + 2) * APITCH + j] = (__bf16)v.z;
        sA[(i4 + 3) * APITCH + j] = (__bf16)v.w;
        if (hasA2) {
          const float4 w = *(const float4*)(ga2 + go);
          sA2[(i4 + 0) * APITCH + j] = (__bf16)w.x;
          sA2[(i4 + 1) * APITCH + j] = (__bf16)w.y;
          sA2[(i4 + 2) * APITCH + j] = (__bf16)w.z;
          sA2[(i4 + 3) * APITCH + j] = (__bf16)w.w;
        }
      }
    } else {
      for (int f = tid; f < GEMM_BM * GEMM_BK; f += 256) {
        int i, j;
        if (kfA) { i = f >> 6; j = f & 63; } else { i = f & (GEMM_BM - 1); j = f >> 7; }
        const int m = m0 + i, k = k0 + j;
        float v = 0.f, v2 = 0.f;
        if (m < p.M && k < p.K) {
          const int off = abase + m * p.sAm + k * p.sAk;
          v = p.A[off];
          if (hasA2) v2 = p.A2[off];
        }
        sA [i * APITCH + j] = (__bf16)v;
        sA2[i * APITCH + j] = (__bf16)v2;
      }
    }
    // ---------------- stage B (+B2), stored [n][k] ----------------
    if (fullN && fullK && kfB) {
      const float* gb  = p.B + n0 * p.sBn + k0;
      const float* gb2 = p.B2 ? (p.B2 + n0 * p.sBn + k0) : nullptr;
#pragma unroll
      for (int f = tid; f < (GEMM_BN * GEMM_BK) / 4; f += 256) {
        const int n = f >> 4, j4 = (f & 15) << 2;
        const int go = n * p.sBn + j4;
        *(v4bf*)(sB + n * APITCH + j4) = cvt4(*(const float4*)(gb + go));
        if (gb2) *(v4bf*)(sB2 + n * APITCH + j4) = cvt4(*(const float4*)(gb2 + go));
      }
    } else if (fullN && fullK && p.sBn == 1) {
      const float* gb  = p.B + n0 + k0 * p.sBk;
      const float* gb2 = p.B2 ? (p.B2 + n0 + k0 * p.sBk) : nullptr;
#pragma unroll
      for (int f = tid; f < (GEMM_BN / 4) * GEMM_BK; f += 256) {
        const int n4 = (f & 15) << 2, j = f >> 4;
        const int go = n4 + j * p.sBk;
        const float4 v = *(const float4*)(gb + go);
        sB[(n4 + 0) * APITCH + j] = (__bf16)v.x;
        sB[(n4 + 1) * APITCH + j] = (__bf16)v.y;
        sB[(n4 + 2) * APITCH + j] = (__bf16)v.z;
        sB[(n4 + 3) * APITCH + j] = (__bf16)v.w;
        if (gb2) {
          const float4 w = *(const float4*)(gb2 + go);
          sB2[(n4 + 0) * APITCH + j] = (__bf16)w.x;
          sB2[(n4 + 1) * APITCH + j] = (__bf16)w.y;
          sB2[(n4 + 2) * APITCH + j] = (__bf16)w.z;
          sB2[(n4 + 3) * APITCH + j] = (__bf16)w.w;
        }
      }
    } else {
      for (int f = tid; f < GEMM_BN * GEMM_BK; f += 256) {
        int n, j;
        if (kfB) { n = f >> 6; j = f & 63; } else { n = f & (GEMM_BN - 1); j = f >> 6; }
        const int nn = n0 + n, k = k0 + j;
        float v = 0.f, v2 = 0.f;
        if (nn < p.N && k < p.K) {
          const int off = k * p.sBk + nn * p.sBn;
          v = p.B[off];
          if (p.B2) v2 = p.B2[off];
        }
        sB [n * APITCH + j] = (__bf16)v;
        sB2[n * APITCH + j] = (__bf16)v2;
      }
    }
    __syncthreads();

    // ---------------- compute: 2 k-steps of 32 per stage ----------------
#pragma unroll
    for (int kk = 0; kk < GEMM_BK; kk += 32) {
      v16bf afr[2], bfr[2];
#pragma unroll
      for (int fm = 0; fm < 2; ++fm)
        afr[fm] = ld_frag(sA + (wm + fm * 16 + l15) * APITCH, kk + kb);
#pragma unroll
      for (int fn = 0; fn < 2; ++fn)
        bfr[fn] = ld_frag(sB + (wn + fn * 16 + l15) * APITCH, kk + kb);
#pragma unroll
      for (int fm = 0; fm < 2; ++fm)
#pragma unroll
        for (int fn = 0; fn < 2; ++fn)
          acc[fm][fn] = wmma_bf16(afr[fm], bfr[fn], acc[fm][fn]);

      if (hasA2) {
        v16bf a2[2], b2[2];
#pragma unroll
        for (int fm = 0; fm < 2; ++fm) {
          a2[fm] = ld_frag(sA2 + (wm + fm * 16 + l15) * APITCH, kk + kb);
          if (neg2) a2[fm] = neg_frag(a2[fm]);
        }
#pragma unroll
        for (int fn = 0; fn < 2; ++fn)
          b2[fn] = ld_frag(sB2 + (wn + fn * 16 + l15) * APITCH, kk + kb);
#pragma unroll
        for (int fm = 0; fm < 2; ++fm)
#pragma unroll
          for (int fn = 0; fn < 2; ++fn)
            acc[fm][fn] = wmma_bf16(a2[fm], b2[fn], acc[fm][fn]);
      }
    }
  }

  // ---- epilogue: C layout lanes0-15 N=lane M=r, lanes16-31 N=lane-16 M=r+8 ----
  const bool doGelu = (p.flags & 1) != 0;
#pragma unroll
  for (int fm = 0; fm < 2; ++fm) {
#pragma unroll
    for (int fn = 0; fn < 2; ++fn) {
      const int n = n0 + wn + fn * 16 + l15;
      if (n >= p.N) continue;
      float badd = 0.f;
      if (p.bias)  badd += p.bias[n];
      if (p.bias2) badd += p.bias2[n];
#pragma unroll
      for (int r = 0; r < 8; ++r) {
        const int m = m0 + wm + fm * 16 + half * 8 + r;
        if (m >= p.M) continue;
        float v = acc[fm][fn][r] + badd;
        if (doGelu) v = gelu_f(v);
        p.C[cbase + m * p.sCm + n * p.sCn] = v;
      }
    }
  }
}

// ---------------------------------------------------------------------------
// Per-mode complex mixing: Q[s] = Z[s](16x128) x W[s](128x128), complex.
// One block per mode s = ky*16+kx (512 modes); 8 waves cover the 128 outputs.
// ---------------------------------------------------------------------------
__global__ __launch_bounds__(256) void modemix_kernel(
    const float* Zr, const float* Zi, float* Qr, float* Qi,
    const float* w1r, const float* w1i, const float* w2r, const float* w2i) {
  __shared__ __attribute__((aligned(16))) __bf16 sAr[16 * 136];
  __shared__ __attribute__((aligned(16))) __bf16 sAi[16 * 136];
  __shared__ __attribute__((aligned(16))) __bf16 sBr[128 * 40];
  __shared__ __attribute__((aligned(16))) __bf16 sBi[128 * 40];

  const int s = blockIdx.x;
  const int ky = s >> 4, kx = s & 15;
  const float* wr; const float* wi; int widx;
  if (ky < 16) { wr = w1r; wi = w1i; widx = ky; }
  else         { wr = w2r; wi = w2i; widx = ky - 16; }
  const int wbase = widx * 16 + kx;  // + i*32768 + o*256

  const int tid = threadIdx.x;
  const float* zr = Zr + s * 2048;
  const float* zi = Zi + s * 2048;
  for (int f = tid; f < 2048; f += 256) {           // A: 16(b) x 128(c)
    const int b = f >> 7, c = f & 127;
    sAr[b * 136 + c] = (__bf16)zr[f];
    sAi[b * 136 + c] = (__bf16)zi[f];
  }

  const int wave = tid >> 5, lane = tid & 31;
  const int half = lane >> 4, l15 = lane & 15;
  const int n0 = wave * 16;
  v8f accr, acci;
#pragma unroll
  for (int r = 0; r < 8; ++r) { accr[r] = 0.f; acci[r] = 0.f; }

  for (int kc = 0; kc < 4; ++kc) {                  // K=128 in chunks of 32
    __syncthreads();
    for (int f = tid; f < 4096; f += 256) {         // B chunk: 32(k) x 128(o)
      const int o = f & 127, j = f >> 7;
      const int goff = (kc * 32 + j) * 32768 + o * 256 + wbase;
      sBr[o * 40 + j] = (__bf16)wr[goff];
      sBi[o * 40 + j] = (__bf16)wi[goff];
    }
    __syncthreads();
    const int ka = kc * 32 + half * 8;
    const int kb = half * 8;
    v16bf ar = ld_frag(sAr + l15 * 136, ka);
    v16bf ai = ld_frag(sAi + l15 * 136, ka);
    v16bf br = ld_frag(sBr + (n0 + l15) * 40, kb);
    v16bf bi = ld_frag(sBi + (n0 + l15) * 40, kb);
    accr = wmma_bf16(ar, br, accr);
    accr = wmma_bf16(neg_frag(ai), bi, accr);       // - Ai*Bi
    acci = wmma_bf16(ar, bi, acci);
    acci = wmma_bf16(ai, br, acci);
  }

  float* qr = Qr + s * 2048;
  float* qi = Qi + s * 2048;
#pragma unroll
  for (int r = 0; r < 8; ++r) {
    const int m = half * 8 + r;                     // batch b
    const int n = n0 + l15;                         // output channel o
    qr[m * 128 + n] = accr[r];
    qi[m * 128 + n] = acci[r];
  }
}

// ---------------------------------------------------------------------------
// DFT matrix generation (tiny).  Kept ky rows: {0..15, 112..127}.
// ---------------------------------------------------------------------------
__global__ __launch_bounds__(256) void gen_dft_kernel(
    float* FH, float* CWm, float* SWm, float* IHC, float* IHS, float* IWC, float* IWS) {
  const int t = blockIdx.x * 256 + threadIdx.x;
  const float TWO_PI = 6.28318530717958647692f;
  if (t < 8192) {                       // FH[h][64]: [cos(ky) | -sin(ky)]
    const int h = t >> 6, j = t & 63, jj = j & 31;
    const int ky = (jj < 16) ? jj : (96 + jj);
    const float ang = TWO_PI * (float)((ky * h) & 127) * (1.0f / 128.0f);
    FH[t] = (j < 32) ? cosf(ang) : -sinf(ang);
  }
  if (t < 2048) {                       // CW/SW[w][kx]
    const int w = t >> 4, kx = t & 15;
    const float ang = TWO_PI * (float)((kx * w) & 127) * (1.0f / 128.0f);
    CWm[t] = cosf(ang);
    SWm[t] = sinf(ang);
  }
  if (t < 4096) {                       // IHC/IHS[j][h], 1/128 folded in
    const int j = t >> 7, h = t & 127;
    const int ky = (j < 16) ? j : (96 + j);
    const float ang = TWO_PI * (float)((ky * h) & 127) * (1.0f / 128.0f);
    IHC[t] = cosf(ang) * (1.0f / 128.0f);
    IHS[t] = sinf(ang) * (1.0f / 128.0f);
  }
  if (t < 2048) {                       // IWC/IWS[kx][w], Hermitian x2 + 1/128
    const int kx = t >> 7, w = t & 127;
    const float a = (kx == 0) ? 1.0f : 2.0f;
    const float ang = TWO_PI * (float)((kx * w) & 127) * (1.0f / 128.0f);
    IWC[t] = a * cosf(ang) * (1.0f / 128.0f);
    IWS[t] = a * sinf(ang) * (1.0f / 128.0f);
  }
}

// Lifting: out[(b,h,w),c] = [x0,x1,gx,gy] . fc0_w[:,c] + fc0_b[c]
__global__ __launch_bounds__(256) void lift_kernel(
    const float* xin, const float* w, const float* bias, float* out) {
  const int t = blockIdx.x * 256 + threadIdx.x;    // 33554432 total
  const int c = t & 127;
  const int m = t >> 7;
  const int ww = m & 127;
  const int h  = (m >> 7) & 127;
  const float x0 = xin[m * 2];
  const float x1 = xin[m * 2 + 1];
  const float gx = (float)h  * (1.0f / 127.0f);
  const float gy = (float)ww * (1.0f / 127.0f);
  out[t] = x0 * w[c] + x1 * w[128 + c] + gx * w[256 + c] + gy * w[384 + c] + bias[c];
}

// Final projection 128 -> 1: one wave per output pixel, shuffle reduction.
__global__ __launch_bounds__(256) void projout_kernel(
    const float* tp, const float* w, const float* b, float* out) {
  const int gid = blockIdx.x * 256 + threadIdx.x;
  const int m = gid >> 5;
  const int lane = threadIdx.x & 31;
  if (m >= 262144) return;
  const float* row = tp + m * 128;
  float sum = 0.f;
  for (int c = lane; c < 128; c += 32) sum += row[c] * w[c];
#pragma unroll
  for (int o = 16; o > 0; o >>= 1) sum += __shfl_down(sum, o, 32);
  if (lane == 0) out[m] = sum + b[0];
}

static void launch_gemm(hipStream_t s, const GemmParams& p, int batch) {
  dim3 grid((p.N + GEMM_BN - 1) / GEMM_BN, (p.M + GEMM_BM - 1) / GEMM_BM, batch);
  gemm_bf16_kernel<<<grid, dim3(256), 0, s>>>(p);
}

extern "C" void kernel_launch(void* const* d_in, const int* in_sizes, int n_in,
                              void* d_out, int out_size, void* d_ws, size_t ws_size,
                              hipStream_t stream) {
  (void)in_sizes; (void)n_in; (void)out_size; (void)ws_size;
  const float* X    = (const float*)d_in[0];
  const float* W1R  = (const float*)d_in[1];
  const float* W1I  = (const float*)d_in[2];
  const float* W2R  = (const float*)d_in[3];
  const float* W2I  = (const float*)d_in[4];
  const float* M1W  = (const float*)d_in[5];
  const float* M1B  = (const float*)d_in[6];
  const float* M2W  = (const float*)d_in[7];
  const float* M2B  = (const float*)d_in[8];
  const float* WSW  = (const float*)d_in[9];
  const float* WSB  = (const float*)d_in[10];
  const float* FC0W = (const float*)d_in[11];
  const float* FC0B = (const float*)d_in[12];
  const float* F1AW = (const float*)d_in[13];
  const float* F1AB = (const float*)d_in[14];
  const float* F1BW = (const float*)d_in[15];
  const float* F1BB = (const float*)d_in[16];
  float* OUT = (float*)d_out;

  float* ws = (float*)d_ws;
  long long off = 0;
  auto alloc = [&](long long n) { float* p = ws + off; off += n; return p; };
  float* ACT0 = alloc(33554432LL);  // activations, (b,h,w,c) fp32
  float* ACT1 = alloc(33554432LL);
  float* X1   = alloc(33554432LL);  // spectral-conv output
  float* T1   = alloc(33554432LL);  // MLP mid / projection mid
  float* Y    = alloc(16777216LL);  // (b, 64=[Yr|Yi], w, c)
  float* ZR   = alloc(1048576LL);   // (s=ky*16+kx, b, c)
  float* ZI   = alloc(1048576LL);
  float* QR   = alloc(1048576LL);
  float* QI   = alloc(1048576LL);
  float* TRr  = alloc(4194304LL);   // (b, c, h, kx)
  float* TIr  = alloc(4194304LL);
  float* FH   = alloc(8192LL);
  float* CWm  = alloc(2048LL);
  float* SWm  = alloc(2048LL);
  float* IHC  = alloc(4096LL);
  float* IHS  = alloc(4096LL);
  float* IWC  = alloc(2048LL);
  float* IWS  = alloc(2048LL);

  gen_dft_kernel<<<32, 256, 0, stream>>>(FH, CWm, SWm, IHC, IHS, IWC, IWS);
  lift_kernel<<<131072, 256, 0, stream>>>(X, FC0W, FC0B, ACT0);

  GemmParams p0{};
  p0.bAdiv = 0x40000000; p0.bCdiv = 0x40000000;

  float* cur = ACT0; float* nxt = ACT1;
  for (int l = 0; l < 4; ++l) {
    // S1: Y[b, 0..63, w, c] = sum_h x[b,h,w,c] * FH[h][j]   (batch = b)
    { GemmParams p = p0;
      p.A = cur; p.sAm = 1; p.sAk = 16384; p.bAlo = 2097152;
      p.B = FH;  p.sBk = 64; p.sBn = 1;
      p.C = Y;   p.sCm = 1; p.sCn = 16384; p.bClo = 1048576;
      p.M = 16384; p.N = 64; p.K = 128;
      launch_gemm(stream, p, 16);
    }
    // S2: Zr = Yr*CW + Yi*SW ; Zi = Yi*CW - Yr*SW   (batch = (b,ky) = 512)
    { GemmParams p = p0;
      p.sAm = 1; p.sAk = 128; p.bAdiv = 32; p.bAhi = 1048576; p.bAlo = 16384;
      p.B = CWm; p.B2 = SWm; p.sBk = 16; p.sBn = 1;
      p.sCm = 1; p.sCn = 2048; p.bCdiv = 32; p.bChi = 128; p.bClo = 32768;
      p.M = 128; p.N = 16; p.K = 128;
      GemmParams pr = p; pr.A = Y;          pr.A2 = Y + 524288; pr.C = ZR; pr.flags = 0;
      launch_gemm(stream, pr, 512);
      GemmParams pi = p; pi.A = Y + 524288; pi.A2 = Y;          pi.C = ZI; pi.flags = 2;
      launch_gemm(stream, pi, 512);
    }
    // S3: per-mode complex mixing with w1/w2
    modemix_kernel<<<512, 256, 0, stream>>>(ZR, ZI, QR, QI,
        W1R + (long long)l * 4194304, W1I + (long long)l * 4194304,
        W2R + (long long)l * 4194304, W2I + (long long)l * 4194304);
    // S4: Tr = Qr*IHC - Qi*IHS ; Ti = Qr*IHS + Qi*IHC   (batch = kx = 16)
    { GemmParams p = p0;
      p.A = QR; p.A2 = QI; p.sAm = 1; p.sAk = 32768; p.bAlo = 2048;
      p.sBk = 128; p.sBn = 1;
      p.sCm = 2048; p.sCn = 16; p.bClo = 1;
      p.M = 2048; p.N = 128; p.K = 32;
      GemmParams pr = p; pr.B = IHC; pr.B2 = IHS; pr.C = TRr; pr.flags = 2;
      launch_gemm(stream, pr, 16);
      GemmParams pi = p; pi.B = IHS; pi.B2 = IHC; pi.C = TIr; pi.flags = 0;
      launch_gemm(stream, pi, 16);
    }
    // S5: x1[b,h,w,c] = Tr*IWC - Ti*IWS   (batch = (b,c) = 2048)
    { GemmParams p = p0;
      p.A = TRr; p.A2 = TIr; p.sAm = 16; p.sAk = 1; p.bAlo = 2048;
      p.B = IWC; p.B2 = IWS; p.sBk = 128; p.sBn = 1;
      p.C = X1; p.sCm = 16384; p.sCn = 128;
      p.bCdiv = 128; p.bChi = 2097152; p.bClo = 1;
      p.M = 128; p.N = 128; p.K = 16; p.flags = 2;
      launch_gemm(stream, p, 2048);
    }
    // S6: T1 = gelu(x1 . mlp1^T + b1)
    { GemmParams p = p0;
      p.A = X1; p.sAm = 128; p.sAk = 1;
      p.B = M1W + l * 16384; p.sBk = 1; p.sBn = 128;
      p.bias = M1B + l * 128;
      p.C = T1; p.sCm = 128; p.sCn = 1;
      p.M = 262144; p.N = 128; p.K = 128; p.flags = 1;
      launch_gemm(stream, p, 1);
    }
    // S7: x_next = [gelu]( x.ws^T + ws_b + T1.mlp2^T + mlp2_b )  (fused dual GEMM)
    { GemmParams p = p0;
      p.A = cur; p.A2 = T1; p.sAm = 128; p.sAk = 1;
      p.B = WSW + l * 16384; p.B2 = M2W + l * 16384;
      p.sBk = 1; p.sBn = 128;
      p.bias = WSB + l * 128; p.bias2 = M2B + l * 128;
      p.C = nxt; p.sCm = 128; p.sCn = 1;
      p.M = 262144; p.N = 128; p.K = 128;
      p.flags = (l < 3) ? 1 : 0;
      launch_gemm(stream, p, 1);
    }
    float* tmp = cur; cur = nxt; nxt = tmp;
  }

  // Projection: T1 = gelu(x . fc1a^T + b); out = T1 . fc1b^T + b
  { GemmParams p = p0;
    p.A = cur; p.sAm = 128; p.sAk = 1;
    p.B = F1AW; p.sBk = 1; p.sBn = 128;
    p.bias = F1AB;
    p.C = T1; p.sCm = 128; p.sCn = 1;
    p.M = 262144; p.N = 128; p.K = 128; p.flags = 1;
    launch_gemm(stream, p, 1);
  }
  projout_kernel<<<32768, 256, 0, stream>>>(T1, F1BW, F1BB, OUT);
}
